// SeparableConvolutionModule_37211596652738
// MI455X (gfx1250) — matrix operand order repaired and checked by reading, not compiled, not run on hardware
//
#include <hip/hip_runtime.h>

typedef __attribute__((ext_vector_type(16))) _Float16 v16h;
typedef __attribute__((ext_vector_type(8)))  _Float16 h8;
typedef __attribute__((ext_vector_type(8)))  float    v8f;

// Problem constants (match reference)
constexpr int Bn = 2, Cn = 3, Hn = 384, Wn = 384, Kn = 51;
constexpr int PH = Hn + Kn - 1;       // 434 padded rows
constexpr int PW = Wn + Kn - 1;       // 434 padded cols
// Tiling
constexpr int STRIP  = 128;           // output pixels per block (x)
constexpr int NSTRIP = Wn / STRIP;    // 3
constexpr int COLS   = STRIP + Kn - 1;// 178 band columns
constexpr int KPAD   = 72;            // taps padded: 144B rows -> 16B-aligned ds_load_b128
constexpr int MROWS  = Cn * (16 + Kn - 1); // 198 rows of the per-tile GEMM
constexpr int MTILES = 13;            // ceil(198/16)

__global__ __launch_bounds__(256)
void sepconv_wmma_kernel(const float* __restrict__ input1,
                         const float* __restrict__ vertical,
                         const float* __restrict__ horizontal,
                         float* __restrict__ out)
{
    // A operand staged transposed as f16: bandT[c][col][i], i-minor for b128 loads
    __shared__ __align__(16) _Float16 bandT[Cn * COLS * KPAD];   // 76,896 B
    __shared__ __align__(16) float    hbuf[Kn * STRIP];          // 26,112 B

    const int blk   = blockIdx.x;
    const int strip = blk % NSTRIP;
    const int y     = (blk / NSTRIP) % Hn;
    const int b     = blk / (NSTRIP * Hn);
    const int x0    = strip * STRIP;
    const int tid   = threadIdx.x;

    // ---- Phase 0a: async-copy horizontal weight tile straight into LDS ----
    // hbuf[j*STRIP+xx] = horizontal[b, j, y, x0+xx]; pure f32 copy -> TDM-free
    // async path (GLOBAL_LOAD_ASYNC_TO_LDS_B128, tracked by ASYNCcnt).
    {
        const float* hsrc = horizontal + ((b * Kn) * Hn + y) * Wn + x0; // + j*Hn*Wn
        unsigned hbase = (unsigned)(unsigned long long)(void*)hbuf;     // LDS offset
        constexpr int CHUNKS = (Kn * STRIP * 4) / 16;                   // 1632 x 16B
        for (int q = tid; q < CHUNKS; q += 256) {
            int j  = q >> 5;              // 32 16B-chunks per 512B row
            int xo = (q & 31) << 4;       // byte offset within row
            unsigned ldsa = hbase + (unsigned)q * 16u;
            unsigned voff = (unsigned)(j * (Hn * Wn * 4) + xo);
            asm volatile("global_load_async_to_lds_b128 %0, %1, %2 offset:0"
                         :: "v"(ldsa), "v"(voff), "s"(hsrc) : "memory");
        }
    }

    // ---- Phase 0b: stage input band (f32 -> f16, transposed) while async runs ----
    {
        int c = 0, i = tid / COLS, col = tid % COLS;  // tid<256 -> c==0
        for (int idx = tid; idx < Cn * Kn * COLS; idx += 256) {
            float v = input1[((b * Cn + c) * PH + (y + i)) * PW + (x0 + col)];
            bandT[(c * COLS + col) * KPAD + i] = (_Float16)v;
            col += 256;
            if (col >= COLS) { col -= COLS; ++i; }
            if (col >= COLS) { col -= COLS; ++i; }
            if (i >= Kn)     { i   -= Kn;  ++c; }
        }
        // zero the K padding tail of each (c,col) row
        for (int row = tid; row < Cn * COLS; row += 256) {
            _Float16* p = &bandT[row * KPAD];
#pragma unroll
            for (int i2 = Kn; i2 < KPAD; ++i2) p[i2] = (_Float16)0.f;
        }
    }

    asm volatile("s_wait_asynccnt 0" ::: "memory");
    __syncthreads();

    // ---- Phase 1: each of the 8 waves owns one 16-pixel x-tile ----
    const int w    = tid >> 5;        // wave id 0..7
    const int lane = tid & 31;
    const int n    = lane & 15;       // pixel within tile / M row within A-tile
    const int half = lane >> 4;       // 0: lanes 0-15, 1: lanes 16-31
    const int xg   = x0 + w * 16 + n; // global output x

    // B operand = vertical weight tile v[i, n], f32->f16, 51 taps padded to 64.
    // ISA B layout (32x16, 16-bit): lane = N, K = half*8+e (e<8), 16+half*8+e-8 per 8-VGPR group.
    const float* vbase = vertical + ((b * Kn) * Hn + y) * Wn + xg; // + i*Hn*Wn
    v16h bf0, bf1;
#pragma unroll
    for (int e = 0; e < 16; ++e) {
        int k0 = (e < 8) ? (half * 8 + e) : (16 + half * 8 + (e - 8));
        int k1 = 32 + k0;
        bf0[e] = (k0 < Kn) ? (_Float16)vbase[k0 * (Hn * Wn)] : (_Float16)0.f;
        bf1[e] = (k1 < Kn) ? (_Float16)vbase[k1 * (Hn * Wn)] : (_Float16)0.f;
    }

    float o0 = 0.f, o1 = 0.f, o2 = 0.f; // out accumulators for c=0,1,2 at pixel n

#pragma unroll
    for (int t = 0; t < MTILES; ++t) {
        // A operand: A[m, i] = bandT[c][w*16 + colL][i],  m = 16t + n
        // ISA A layout (16x32, 16-bit): lane = M; K = half*8+e (e<8), 16+half*8+e-8
        int  mA  = t * 16 + n;
        bool av  = (t < MTILES - 1) | (mA < MROWS);   // folds for t<12
        int  mac = av ? mA : 0;
        int  cA  = (mac >= 132) ? 2 : ((mac >= 66) ? 1 : 0);   // no division
        int  colL = mac - cA * 66;
        const _Float16* rp = &bandT[(cA * COLS + (w * 16 + colL)) * KPAD];

        h8 p0 = *(const h8*)(rp      + half * 8);   // K 0..7   (+half*8)
        h8 p1 = *(const h8*)(rp + 16 + half * 8);   // K 16..23 (+half*8)
        h8 p2 = *(const h8*)(rp + 32 + half * 8);   // K 32..39 (+half*8)
        h8 p3 = *(const h8*)(rp + 48 + half * 8);   // K 48..55 (+half*8)
        if (!av) { h8 z = {}; p0 = z; p1 = z; p2 = z; p3 = z; }

        v16h a0, a1;
#pragma unroll
        for (int e = 0; e < 8; ++e) {
            a0[e] = p0[e]; a0[e + 8] = p1[e];
            a1[e] = p2[e]; a1[e + 8] = p3[e];
        }

        // D[(c,colL), n] = sum_i band * vertical   (f32 accumulate)
        v8f acc = {};
        acc = __builtin_amdgcn_wmma_f32_16x16x32_f16(false, a0, false, bf0,
                                                     (short)0, acc, false, false);
        acc = __builtin_amdgcn_wmma_f32_16x16x32_f16(false, a1, false, bf1,
                                                     (short)0, acc, false, false);

        // Stage 2: banded-diagonal contraction with horizontal weights.
        // C/D layout: lane holds N = lane&15; VGPR r holds M = 16t + r + 8*half.
#pragma unroll
        for (int r = 0; r < 8; ++r) {
            const int mD = t * 16 + half * 8 + r;   // half-uniform; const after unroll
            if (mD < MROWS) {
                const int cD   = (mD >= 132) ? 2 : ((mD >= 66) ? 1 : 0); // const
                const int base = mD - cD * 66;                            // const
                int  jl = base - n;                 // horizontal tap index, per-lane
                bool jv = (jl >= 0) && (jl < Kn);
                int  jc = jv ? jl : 0;
                float hv = hbuf[jc * STRIP + w * 16 + n];
                hv = jv ? hv : 0.f;
                float prod = acc[r] * hv;
                if (cD == 0)      o0 += prod;       // folds to direct add
                else if (cD == 1) o1 += prod;
                else              o2 += prod;
            }
        }
    }

    // Halves hold complementary M ranges of the same pixel n: combine across lane^16.
    o0 += __shfl_xor(o0, 16, 32);
    o1 += __shfl_xor(o1, 16, 32);
    o2 += __shfl_xor(o2, 16, 32);
    if (half == 0) {
        out[((b * Cn + 0) * Hn + y) * Wn + xg] = o0;
        out[((b * Cn + 1) * Hn + y) * Wn + xg] = o1;
        out[((b * Cn + 2) * Hn + y) * Wn + xg] = o2;
    }
}

extern "C" void kernel_launch(void* const* d_in, const int* in_sizes, int n_in,
                              void* d_out, int out_size, void* d_ws, size_t ws_size,
                              hipStream_t stream) {
    (void)in_sizes; (void)n_in; (void)d_ws; (void)ws_size; (void)out_size;
    const float* input1     = (const float*)d_in[0];
    const float* vertical   = (const float*)d_in[1];
    const float* horizontal = (const float*)d_in[2];
    float* out              = (float*)d_out;

    dim3 grid(Bn * Hn * NSTRIP);   // 2304 blocks, one (b, y, 128-px strip) each
    dim3 block(256);               // 8 waves, one 16-pixel WMMA tile per wave
    sepconv_wmma_kernel<<<grid, block, 0, stream>>>(input1, vertical, horizontal, out);
}
